// QuantizedLinear_23519240912995
// MI455X (gfx1250) — matrix operand order repaired and checked by reading, not compiled, not run on hardware
//
#include <hip/hip_runtime.h>

typedef __attribute__((ext_vector_type(16))) _Float16 v16h;
typedef __attribute__((ext_vector_type(8)))  _Float16 v8h;
typedef __attribute__((ext_vector_type(4)))  _Float16 v4h;
typedef __attribute__((ext_vector_type(2)))  _Float16 v2h;
typedef __attribute__((ext_vector_type(8)))  float    v8f;

#define O_DIM 11008
#define I_DIM 4096
#define M_DIM 8192
#define BM 128
#define BN 128
#define BK 32
#define KTILES (I_DIM / BK)   // 128
#define NSCALE (I_DIM / 128)  // 32 scales per output row

__global__ __launch_bounds__(256) void qlin_wmma_kernel(
    const float*    __restrict__ x,       // [8192, 4096] f32
    const int*      __restrict__ pw,      // [11008, 2048] int32, 1 byte each (2 nibbles)
    const _Float16* __restrict__ scales,  // [11008 * 32] f16
    const float*    __restrict__ bias,    // [11008]
    float*          __restrict__ out)     // [8192, 11008]
{
    // Double-buffered staging.
    // A: x tile as f16, row-major [M][K] -> A-fragment = 2x contiguous b128.
    __shared__ __align__(16) _Float16 Alds[2][BM][BK];            // 2 x 8 KB
    // B: fragment-ready layout [n-tile][kHalf][col][pair]: each lane's 16 f16
    // for one WMMA B operand are 32 contiguous bytes -> 2x ds_load_b128.
    __shared__ __align__(16) v2h      Blds[2][BN / 16][2][16][8]; // 2 x 8 KB

    const int tid  = threadIdx.x;
    const int lane = tid & 31;
    const int wid  = tid >> 5;

    const int nTile = blockIdx.x * BN;
    const int mTile = blockIdx.y * BM;

    // wave sub-tile: 32 rows x 64 cols -> 2x4 WMMA tiles
    const int waveM = (wid >> 1) * 32;
    const int waveN = (wid & 1) * 64;

    v8f acc[2][4];
    const v8f vzero = {0.f, 0.f, 0.f, 0.f, 0.f, 0.f, 0.f, 0.f};
#pragma unroll
    for (int i = 0; i < 2; ++i)
#pragma unroll
        for (int j = 0; j < 4; ++j) acc[i][j] = vzero;

    // X loader mapping: thread covers 4 rows, 4 consecutive f32 each
    const int xr0 = tid >> 3;         // rows xr0 + 32*i
    const int xc  = (tid & 7) * 4;    // 0..28
    // W loader mapping: 2 threads per output row, 8 int32 (=16 weights) each
    const int wn  = tid >> 1;         // 0..127
    const int wp0 = (tid & 1) * 8;    // 0 or 8 (pair offset within BK/2=16)

    const int    oGlob    = nTile + wn;
    const size_t wRowBase = (size_t)oGlob * (I_DIM / 2);

    // ---- stage tile kt into buffer buf ----
    auto stage = [&](int kt, int buf) {
        const int k0 = kt * BK;
        // x tile: f32 -> f16
#pragma unroll
        for (int i = 0; i < 4; ++i) {
            const int r = xr0 + 32 * i;
            const float4 v = *(const float4*)&x[(size_t)(mTile + r) * I_DIM + k0 + xc];
            v4h h;
            h.x = (_Float16)v.x; h.y = (_Float16)v.y;
            h.z = (_Float16)v.z; h.w = (_Float16)v.w;
            *(v4h*)&Alds[buf][r][xc] = h;
        }
        // weight tile: unpack 4-bit, scale, write fragment-ready 32B run
        const float s    = (float)scales[(size_t)oGlob * NSCALE + (k0 >> 7)];
        const int* wrow  = &pw[wRowBase + (k0 >> 1) + wp0];
        v8h w0, w1;
#pragma unroll
        for (int p = 0; p < 4; ++p) {
            const int q0 = wrow[p];
            w0[2 * p]     = (_Float16)(s * (float)((q0 & 15) - 7));
            w0[2 * p + 1] = (_Float16)(s * (float)(((q0 >> 4) & 15) - 7));
            const int q1 = wrow[p + 4];
            w1[2 * p]     = (_Float16)(s * (float)((q1 & 15) - 7));
            w1[2 * p + 1] = (_Float16)(s * (float)(((q1 >> 4) & 15) - 7));
        }
        v8h* dst = (v8h*)&Blds[buf][wn >> 4][wp0 >> 3][wn & 15][0];
        dst[0] = w0;
        dst[1] = w1;
    };

    // ---- consume tile in buffer buf: 8x v_wmma_f32_16x16x32_f16 ----
    auto compute = [&](int buf) {
        v16h aFrag[2];
#pragma unroll
        for (int mi = 0; mi < 2; ++mi) {
            const int row   = waveM + mi * 16 + (lane & 15);
            const int khalf = (lane >> 4) * 16;  // lanes 0-15: K 0-15, 16-31: K 16-31
            v8h lo = *(const v8h*)&Alds[buf][row][khalf];
            v8h hi = *(const v8h*)&Alds[buf][row][khalf + 8];
            aFrag[mi] = __builtin_shufflevector(lo, hi,
                0, 1, 2, 3, 4, 5, 6, 7, 8, 9, 10, 11, 12, 13, 14, 15);
        }
        v16h bFrag[4];
#pragma unroll
        for (int ni = 0; ni < 4; ++ni) {
            const v2h* bp = &Blds[buf][(waveN >> 4) + ni][lane >> 4][lane & 15][0];
            v8h lo = *(const v8h*)bp;        // pairs 0..3 -> K 0..7 of this half
            v8h hi = *(const v8h*)(bp + 4);  // pairs 4..7 -> K 8..15
            bFrag[ni] = __builtin_shufflevector(lo, hi,
                0, 1, 2, 3, 4, 5, 6, 7, 8, 9, 10, 11, 12, 13, 14, 15);
        }
#pragma unroll
        for (int mi = 0; mi < 2; ++mi)
#pragma unroll
            for (int ni = 0; ni < 4; ++ni)
                acc[mi][ni] = __builtin_amdgcn_wmma_f32_16x16x32_f16(
                    false, aFrag[mi], false, bFrag[ni],
                    (short)0, acc[mi][ni], false, false);
    };

    stage(0, 0);
    __syncthreads();

    for (int kt = 0; kt < KTILES; ++kt) {
        const int cur = kt & 1;
        if (kt + 1 < KTILES) {
            stage(kt + 1, cur ^ 1);
            if (kt + 2 < KTILES) {
                const int k2 = (kt + 2) * BK;
                __builtin_prefetch(&x[(size_t)(mTile + xr0) * I_DIM + k2 + xc], 0, 1);
                __builtin_prefetch(&pw[wRowBase + (k2 >> 1) + wp0], 0, 1);
            }
        }
        compute(cur);
        __syncthreads();
    }

    // ---- epilogue: C layout (VGPR j -> M=j / M=8+j, N=lane%16), add bias ----
#pragma unroll
    for (int mi = 0; mi < 2; ++mi) {
#pragma unroll
        for (int ni = 0; ni < 4; ++ni) {
            const int col  = nTile + waveN + ni * 16 + (lane & 15);
            const float bc = bias[col];
            const int rb   = mTile + waveM + mi * 16 + ((lane >> 4) << 3);
#pragma unroll
            for (int j = 0; j < 8; ++j) {
                out[(size_t)(rb + j) * O_DIM + col] = acc[mi][ni][j] + bc;
            }
        }
    }
}

extern "C" void kernel_launch(void* const* d_in, const int* in_sizes, int n_in,
                              void* d_out, int out_size, void* d_ws, size_t ws_size,
                              hipStream_t stream) {
    const float*    x      = (const float*)d_in[0];
    const int*      pw     = (const int*)d_in[1];
    const _Float16* scales = (const _Float16*)d_in[2];
    const float*    bias   = (const float*)d_in[3];
    float*          out    = (float*)d_out;

    dim3 grid(O_DIM / BN, M_DIM / BM);   // (86, 64)
    dim3 block(256);
    qlin_wmma_kernel<<<grid, block, 0, stream>>>(x, pw, scales, bias, out);
}
